// GCNBlockTimeInv_75230647157513
// MI455X (gfx1250) — compile-verified, matrix operand order checked by reading, not run on hardware
//
#include <hip/hip_runtime.h>

#define FEAT 64

typedef __attribute__((ext_vector_type(16))) __bf16 v16bf;
typedef __attribute__((ext_vector_type(8)))  float  v8f;

// ---------------- degree / normalization ----------------

__global__ void deg_init_kernel(float* __restrict__ deg, int N) {
    int i = blockIdx.x * 256 + threadIdx.x;
    if (i < N) deg[i] = 1.0f;  // self-loop contributes 1
}

__global__ void deg_count_kernel(const int* __restrict__ col, float* __restrict__ deg, int E) {
    int e = blockIdx.x * 256 + threadIdx.x;
    if (e < E) unsafeAtomicAdd(&deg[col[e]], 1.0f);
}

__global__ void deg_rsqrt_kernel(float* __restrict__ deg, int N) {
    int i = blockIdx.x * 256 + threadIdx.x;
    if (i < N) deg[i] = rsqrtf(deg[i]);
}

// ---------------- W packing: f32 [64x64] -> bf16 WMMA B-fragments ----------------
// Fragment fi = kc*4 + t covers K = kc*32..kc*32+31, cols t*16..t*16+15.
// Lane l holds 16 bf16: koff = (l>>4)*8, elem j<8 -> K = kc*32+koff+j,
// elem j>=8 -> K = kc*32+koff+8+j. Stored contiguously: Wp[fi*512 + l*16 + j].

__global__ void pack_w_kernel(const float* __restrict__ Wm, __bf16* __restrict__ Wp) {
    int tid = blockIdx.x * 256 + threadIdx.x;
    if (tid >= 8 * 32 * 16) return;
    int fi = tid >> 9;
    int l  = (tid >> 4) & 31;
    int j  = tid & 15;
    int kc = fi >> 2;
    int t  = fi & 3;
    int koff = (l >> 4) << 3;
    int K   = kc * 32 + koff + ((j < 8) ? j : (8 + j));
    int col = t * 16 + (l & 15);
    Wp[tid] = (__bf16)Wm[K * FEAT + col];
}

// ---------------- GEMM: H = X @ W via bf16 WMMA ----------------
// One wave sweeps 4 row tiles (64 rows x 64 cols), holding all 8 packed
// B fragments in VGPRs for the whole sweep. 8 WMMAs per tile.
// All tile/row-base arithmetic forced scalar via readfirstlane so bounds
// tests lower to s_cmp/s_cbranch instead of EXEC-mask dances.

__global__ __launch_bounds__(128) void gemm_bf16_kernel(const float* __restrict__ X,
                                                        const __bf16* __restrict__ Wp,
                                                        float* __restrict__ H, int N) {
    const int lane     = threadIdx.x & 31;
    const int waveId   = __builtin_amdgcn_readfirstlane(threadIdx.x >> 5);  // SGPR
    const int numTiles = (N + 15) >> 4;
    const int tile0    = (blockIdx.x * 4 + waveId) * 4;                     // scalar
    if (tile0 >= numTiles) return;            // scalar branch, uniform exit

    // load packed B fragments once: 16 x global_load_b128 total
    v16bf bfrag[8];
#pragma unroll
    for (int fi = 0; fi < 8; ++fi)
        bfrag[fi] = *(const v16bf*)(Wp + fi * 512 + lane * 16);

    const int m    = lane & 15;
    const int koff = (lane >> 4) << 3;        // 0 or 8
    const int n    = lane & 15;
    const int mb   = (lane >> 4) << 3;

    for (int tt = 0; tt < 4; ++tt) {
        const int tile = tile0 + tt;          // scalar
        if (tile >= numTiles) break;          // scalar branch
        const int rowBase = tile << 4;        // scalar
        const int row = min(rowBase + m, N - 1);   // branch-free per-lane clamp

        // A fragments: 8 x b128 loads, convert f32->bf16 in regs
        v16bf a[2];
#pragma unroll
        for (int kc = 0; kc < 2; ++kc) {
            const float* xp = X + (size_t)row * FEAT + kc * 32 + koff;
            float4 p0 = *(const float4*)(xp);
            float4 p1 = *(const float4*)(xp + 4);
            float4 p2 = *(const float4*)(xp + 16);
            float4 p3 = *(const float4*)(xp + 20);
            a[kc][0]  = (__bf16)p0.x;  a[kc][1]  = (__bf16)p0.y;
            a[kc][2]  = (__bf16)p0.z;  a[kc][3]  = (__bf16)p0.w;
            a[kc][4]  = (__bf16)p1.x;  a[kc][5]  = (__bf16)p1.y;
            a[kc][6]  = (__bf16)p1.z;  a[kc][7]  = (__bf16)p1.w;
            a[kc][8]  = (__bf16)p2.x;  a[kc][9]  = (__bf16)p2.y;
            a[kc][10] = (__bf16)p2.z;  a[kc][11] = (__bf16)p2.w;
            a[kc][12] = (__bf16)p3.x;  a[kc][13] = (__bf16)p3.y;
            a[kc][14] = (__bf16)p3.z;  a[kc][15] = (__bf16)p3.w;
        }

        v8f acc[4] = {};
#pragma unroll
        for (int t = 0; t < 4; ++t) {
#pragma unroll
            for (int kc = 0; kc < 2; ++kc) {
                acc[t] = __builtin_amdgcn_wmma_f32_16x16x32_bf16(
                    false, a[kc], false, bfrag[kc * 4 + t], (short)0, acc[t], false, false);
            }
        }

        // D layout: VGPR r -> lanes 0-15: M=r, N=lane; lanes 16-31: M=8+r, N=lane-16
        float* hp = H + (size_t)(rowBase + mb) * FEAT + n;
        if (rowBase + 16 <= N) {              // scalar branch; always taken when N%16==0
#pragma unroll
            for (int t = 0; t < 4; ++t) {
#pragma unroll
                for (int r = 0; r < 8; ++r) {
                    hp[(size_t)r * FEAT + t * 16] = acc[t][r];
                }
            }
        } else {
            // partial last tile (cold path): per-row guard
#pragma unroll
            for (int t = 0; t < 4; ++t) {
#pragma unroll
                for (int r = 0; r < 8; ++r) {
                    if (rowBase + mb + r < N) hp[(size_t)r * FEAT + t * 16] = acc[t][r];
                }
            }
        }
    }
}

// ---------------- edge scatter: agg[col] += H[row] * dinv[row]*dinv[col] ----------------
// 16 threads per edge, float4 per thread: 256B coalesced gather + packed f32 atomics
// landing in the 192MB L2 (whole working set is L2-resident).

__global__ void scatter_kernel(const float* __restrict__ H,
                               const int* __restrict__ row,
                               const int* __restrict__ col,
                               const float* __restrict__ dinv,
                               float* __restrict__ agg, int E16) {
    int t = blockIdx.x * 256 + threadIdx.x;
    if (t >= E16) return;
    int e = t >> 4;
    int f = (t & 15) << 2;
    int r = row[e];
    int c = col[e];
    float nrm = dinv[r] * dinv[c];
    float4 h  = *(const float4*)(H + (size_t)r * FEAT + f);
    float* ap = agg + (size_t)c * FEAT + f;
    unsafeAtomicAdd(ap + 0, h.x * nrm);
    unsafeAtomicAdd(ap + 1, h.y * nrm);
    unsafeAtomicAdd(ap + 2, h.z * nrm);
    unsafeAtomicAdd(ap + 3, h.w * nrm);
}

// ---------------- epilogue: x = relu(agg + self_loop + b), in place ----------------

__global__ void epilogue_kernel(const float* __restrict__ H,
                                const float* __restrict__ dinv,
                                const float* __restrict__ bias,
                                float* __restrict__ agg, int N16) {
    int t = blockIdx.x * 256 + threadIdx.x;
    if (t >= N16) return;
    int n = t >> 4;
    int f = (t & 15) << 2;
    float d2 = dinv[n];
    d2 *= d2;                                   // self-loop norm = dinv[i]^2
    float4 h  = *(const float4*)(H + (size_t)n * FEAT + f);
    float4 a  = *(const float4*)(agg + (size_t)n * FEAT + f);
    float4 bb = *(const float4*)(bias + f);
    float4 o;
    o.x = fmaxf(a.x + h.x * d2 + bb.x, 0.0f);
    o.y = fmaxf(a.y + h.y * d2 + bb.y, 0.0f);
    o.z = fmaxf(a.z + h.z * d2 + bb.z, 0.0f);
    o.w = fmaxf(a.w + h.w * d2 + bb.w, 0.0f);
    *(float4*)(agg + (size_t)n * FEAT + f) = o;
}

// ---------------- launch ----------------

extern "C" void kernel_launch(void* const* d_in, const int* in_sizes, int n_in,
                              void* d_out, int out_size, void* d_ws, size_t ws_size,
                              hipStream_t stream) {
    (void)n_in; (void)out_size; (void)ws_size;
    const float* x  = (const float*)d_in[0];
    const int*   ei = (const int*)d_in[1];    // [2, E] flattened: sources then targets
    const float* Wm = (const float*)d_in[4];
    const float* b  = (const float*)d_in[5];

    const int N = in_sizes[0] / FEAT;
    const int E = in_sizes[1] / 2;
    const int* row = ei;         // sources (gather side)
    const int* col = ei + E;     // targets (aggregation side)

    char*  ws  = (char*)d_ws;
    size_t off = 0;
    const size_t nodeBytes = (size_t)N * FEAT * sizeof(float);
    float*  dinv  = (float*)(ws + off);  off += ((size_t)N * 4 + 255) & ~(size_t)255;
    __bf16* Wpack = (__bf16*)(ws + off); off += (8 * 512 * sizeof(__bf16) + 255) & ~(size_t)255;
    float*  Hbuf  = (float*)(ws + off);  off += (nodeBytes + 255) & ~(size_t)255;
    float*  buf0  = (float*)(ws + off);  off += (nodeBytes + 255) & ~(size_t)255;
    float*  buf1  = (float*)(ws + off);  off += (nodeBytes + 255) & ~(size_t)255;

    // one-time: normalization terms + packed bf16 W (shared across all 4 layers)
    deg_init_kernel <<<(N + 255) / 256, 256, 0, stream>>>(dinv, N);
    deg_count_kernel<<<(E + 255) / 256, 256, 0, stream>>>(col, dinv, E);
    deg_rsqrt_kernel<<<(N + 255) / 256, 256, 0, stream>>>(dinv, N);
    pack_w_kernel   <<<16, 256, 0, stream>>>(Wm, Wpack);

    const int numTiles   = (N + 15) / 16;
    const int wavesNeed  = (numTiles + 3) / 4;      // 4 tiles per wave
    const int gemmBlocks = (wavesNeed + 3) / 4;     // 4 waves per block
    const int E16 = E * 16;
    const int N16 = N * 16;

    const float* xcur = x;
    for (int l = 0; l < 4; ++l) {
        float* agg = (l == 3) ? (float*)d_out : ((l & 1) ? buf1 : buf0);
        gemm_bf16_kernel<<<gemmBlocks, 128, 0, stream>>>(xcur, Wpack, Hbuf, N);
        hipMemsetAsync(agg, 0, nodeBytes, stream);
        scatter_kernel <<<(E16 + 255) / 256, 256, 0, stream>>>(Hbuf, row, col, dinv, agg, E16);
        epilogue_kernel<<<(N16 + 255) / 256, 256, 0, stream>>>(Hbuf, dinv, b, agg, N16);
        xcur = agg;
    }
}
